// AdvancedNeuralMemory_74964359185049
// MI455X (gfx1250) — compile-verified, matrix operand order checked by reading, not run on hardware
//
#include <hip/hip_runtime.h>

// ---------------------------------------------------------------------------
// Problem constants (match reference)
// ---------------------------------------------------------------------------
static constexpr int Bb   = 4;
static constexpr int Ss   = 8192;
static constexpr int Dd   = 1024;
static constexpr int Mm   = 512;
static constexpr int ROWS = Bb * Ss;          // 32768
static constexpr int CHUNKS = 64;             // scan chunks along S
static constexpr int CLEN   = 128;            // chunk length (64*128 = 8192)

typedef __bf16 bf16_t;
typedef bf16_t v16bf __attribute__((ext_vector_type(16)));
typedef bf16_t v8bf  __attribute__((ext_vector_type(8)));
typedef float  v8f   __attribute__((ext_vector_type(8)));
typedef __attribute__((address_space(3))) bf16_t lds_bf16;

// ---------------------------------------------------------------------------
// Helpers
// ---------------------------------------------------------------------------
__device__ __forceinline__ unsigned bf_bits(float f) {
    union { float f; unsigned u; } c; c.f = f;
    unsigned u = c.u;
    return (u + 0x7FFFu + ((u >> 16) & 1u)) >> 16;   // round-to-nearest-even
}
__device__ __forceinline__ unsigned bf_pack2(float a, float b) {
    return bf_bits(a) | (bf_bits(b) << 16);
}
__device__ __forceinline__ float gelu_tanh(float x) {
    float x3 = x * x * x;
    float t  = tanhf(0.7978845608028654f * (x + 0.044715f * x3));
    return 0.5f * x * (1.0f + t);
}
__device__ __forceinline__ float sigmoidf_dev(float x) {
    return 1.0f / (1.0f + expf(-x));
}
// wave-relative LDS byte address (for async-to-LDS dsaddr operand)
__device__ __forceinline__ unsigned lds_addr_of(const bf16_t* p) {
    return (unsigned)(unsigned long long)(lds_bf16*)p;
}

// ---------------------------------------------------------------------------
// Weight convert + transpose: Wt[n*K + k] = bf16(W[k*N + n])
// ---------------------------------------------------------------------------
__global__ void convert_transpose_kernel(const float* __restrict__ W,
                                         unsigned short* __restrict__ Wt,
                                         int K, int N) {
    int idx = blockIdx.x * blockDim.x + threadIdx.x;
    if (idx >= K * N) return;
    int k = idx / N;
    int n = idx - k * N;
    Wt[(size_t)n * K + k] = (unsigned short)bf_bits(W[idx]);
}

// ---------------------------------------------------------------------------
// WMMA GEMM:  out[rows,N] = epilogue( A[rows,K](f32) x W[K,N](bf16, stored
// transposed as Wt[N][K]) )
// Block tile 128x128x64, 256 threads (8 wave32), wave tile 64x32 (4x2 WMMAs).
// Double-buffered LDS pipeline; B tile staged with GLOBAL_LOAD_ASYNC_TO_LDS
// (ASYNCcnt), A tile staged through VGPRs with in-flight f32->bf16 convert.
// ---------------------------------------------------------------------------
#define GBM 128
#define GBN 128
#define GBK 64
#define LDSS 80   // padded LDS row stride in bf16 elems (160B, 16B-multiple)

enum { MODE_BIAS = 0, MODE_GELU = 1, MODE_U = 2, MODE_RESID = 3 };

template <int MODE>
__global__ __launch_bounds__(256)
void gemm_bf16_wmma(const float* __restrict__ A,
                    const unsigned short* __restrict__ Wt,
                    const float* __restrict__ bias,
                    float* __restrict__ out,
                    int K, int N,
                    const float* __restrict__ resid,
                    const float* __restrict__ scal) {
    __shared__ bf16_t As[2][GBM * LDSS];
    __shared__ bf16_t Bs[2][GBN * LDSS];

    const int tid  = threadIdx.x;
    const int lane = tid & 31;
    const int wv   = tid >> 5;        // 0..7
    const int wrow = wv >> 2;         // 0..1  -> 64-row slab
    const int wcol = wv & 3;          // 0..3  -> 32-col slab
    const int lhalf = lane >> 4;      // 0/1
    const int lr16  = lane & 15;

    const size_t rm0 = (size_t)blockIdx.y * GBM;
    const int    n0  = blockIdx.x * GBN;

    v8f acc[4][2];
#pragma unroll
    for (int mt = 0; mt < 4; ++mt)
#pragma unroll
        for (int nt = 0; nt < 2; ++nt)
#pragma unroll
            for (int r = 0; r < 8; ++r) acc[mt][nt][r] = 0.0f;

    float4 areg[8];

    // ---- stage helpers ----------------------------------------------------
    // B tile: 128x64 bf16 via async copy straight into LDS (raw bytes).
    auto issueB = [&](int buf, int k0) {
#pragma unroll
        for (int i = 0; i < 4; ++i) {
            int idx = tid + i * 256;          // 0..1023
            int r   = idx >> 3;               // 0..127
            int c   = idx & 7;                // 0..7  (16B chunk)
            const unsigned short* g = Wt + (size_t)(n0 + r) * K + k0 + c * 8;
            unsigned dsaddr = lds_addr_of(&Bs[buf][r * LDSS + c * 8]);
            asm volatile("global_load_async_to_lds_b128 %0, %1, off"
                         :: "v"(dsaddr), "v"(g) : "memory");
        }
    };
    // A tile: issue 128x64 f32 global loads into registers.
    auto loadA = [&](int k0) {
#pragma unroll
        for (int i = 0; i < 8; ++i) {
            int idx = tid + i * 256;          // 0..2047
            int r   = idx >> 4;               // 0..127
            int c4  = idx & 15;               // 0..15 (float4 column)
            areg[i] = *(const float4*)(A + (rm0 + r) * (size_t)K + k0 + c4 * 4);
        }
    };
    // Convert staged A registers to bf16 and store to LDS buffer.
    auto storeA = [&](int buf) {
#pragma unroll
        for (int i = 0; i < 8; ++i) {
            int idx = tid + i * 256;
            int r   = idx >> 4;
            int c4  = idx & 15;
            unsigned* dst = (unsigned*)&As[buf][r * LDSS + c4 * 4];
            dst[0] = bf_pack2(areg[i].x, areg[i].y);
            dst[1] = bf_pack2(areg[i].z, areg[i].w);
        }
    };
    // WMMA compute on one resident LDS buffer (two K=32 steps).
    auto compute = [&](int buf) {
#pragma unroll
        for (int kk = 0; kk < GBK; kk += 32) {
            v16bf af[4], bfr[2];
#pragma unroll
            for (int mt = 0; mt < 4; ++mt) {
                const bf16_t* p =
                    &As[buf][(wrow * 64 + mt * 16 + lr16) * LDSS + kk + lhalf * 8];
                v8bf lo = *(const v8bf*)p;
                v8bf hi = *(const v8bf*)(p + 16);
                af[mt] = __builtin_shufflevector(lo, hi,
                         0, 1, 2, 3, 4, 5, 6, 7, 8, 9, 10, 11, 12, 13, 14, 15);
            }
#pragma unroll
            for (int nt = 0; nt < 2; ++nt) {
                const bf16_t* p =
                    &Bs[buf][(wcol * 32 + nt * 16 + lr16) * LDSS + kk + lhalf * 8];
                v8bf lo = *(const v8bf*)p;
                v8bf hi = *(const v8bf*)(p + 16);
                bfr[nt] = __builtin_shufflevector(lo, hi,
                          0, 1, 2, 3, 4, 5, 6, 7, 8, 9, 10, 11, 12, 13, 14, 15);
            }
#pragma unroll
            for (int mt = 0; mt < 4; ++mt)
#pragma unroll
                for (int nt = 0; nt < 2; ++nt)
                    acc[mt][nt] = __builtin_amdgcn_wmma_f32_16x16x32_bf16(
                        false, af[mt], false, bfr[nt], (short)0, acc[mt][nt],
                        false, false);
        }
    };

    // ---- prologue: stage tile 0 into buffer 0 -----------------------------
    issueB(0, 0);
    loadA(0);
    storeA(0);
    asm volatile("s_wait_asynccnt 0x0" ::: "memory");
    __syncthreads();

    // ---- main pipelined loop ---------------------------------------------
    const int nTiles = K / GBK;
    for (int t = 0; t < nTiles; ++t) {
        const int cur = t & 1;
        const int nxt = cur ^ 1;
        const bool has_next = (t + 1) < nTiles;
        if (has_next) {
            issueB(nxt, (t + 1) * GBK);   // async bytes -> LDS (ASYNCcnt)
            loadA((t + 1) * GBK);         // f32 -> VGPRs (LOADcnt)
        }
        compute(cur);                     // WMMAs hide the staging latency
        if (has_next) {
            storeA(nxt);                  // convert + ds_store into next buf
            asm volatile("s_wait_asynccnt 0x0" ::: "memory");
        }
        __syncthreads();                  // next buffer ready for all waves
    }

    // ---- epilogue ---------------------------------------------------------
    float lrv = 0.0f;
    if (MODE == MODE_U) lrv = *scal;

#pragma unroll
    for (int mt = 0; mt < 4; ++mt) {
#pragma unroll
        for (int nt = 0; nt < 2; ++nt) {
            const size_t row0 = rm0 + (size_t)(wrow * 64 + mt * 16 + lhalf * 8);
            const int    col  = n0 + wcol * 32 + nt * 16 + lr16;
            float bcol = 0.0f;
            if (MODE == MODE_BIAS || MODE == MODE_RESID) bcol = bias[col];
#pragma unroll
            for (int r = 0; r < 8; ++r) {
                const size_t o = (row0 + r) * (size_t)N + col;
                float val = acc[mt][nt][r];
                if (MODE == MODE_BIAS)       out[o] = val + bcol;
                else if (MODE == MODE_GELU)  out[o] = gelu_tanh(val);
                else if (MODE == MODE_U)     out[o] = lrv * (resid[o] - val);
                else /* MODE_RESID */        out[o] = val + bcol + resid[o];
            }
        }
    }
}

// ---------------------------------------------------------------------------
// LayerNorm over rows of length 512 (in place)
// ---------------------------------------------------------------------------
__global__ __launch_bounds__(256)
void layernorm_kernel(float* __restrict__ buf,
                      const float* __restrict__ gamma,
                      const float* __restrict__ beta) {
    __shared__ float s1[256];
    __shared__ float s2[256];
    const size_t row = blockIdx.x;
    const int t = threadIdx.x;
    float a = buf[row * Mm + t];
    float b = buf[row * Mm + t + 256];
    s1[t] = a + b;
    s2[t] = a * a + b * b;
    __syncthreads();
    for (int off = 128; off > 0; off >>= 1) {
        if (t < off) { s1[t] += s1[t + off]; s2[t] += s2[t + off]; }
        __syncthreads();
    }
    const float mean = s1[0] * (1.0f / Mm);
    const float var  = s2[0] * (1.0f / Mm) - mean * mean;
    const float inv  = rsqrtf(var + 1e-5f);
    buf[row * Mm + t]       = (a - mean) * inv * gamma[t] + beta[t];
    buf[row * Mm + t + 256] = (b - mean) * inv * gamma[t + 256] + beta[t + 256];
}

// ---------------------------------------------------------------------------
// Blocked linear scan: mem_t = g*mem_{t-1} + u_t   (g scalar = sigmoid(ff))
// Pass A: local inclusive scan per chunk + chunk carries (in place on u)
// ---------------------------------------------------------------------------
__global__ __launch_bounds__(512)
void scan_local_kernel(float* __restrict__ u, float* __restrict__ carry,
                       const float* __restrict__ ff) {
    const int bc = blockIdx.x;           // b*CHUNKS + c
    const int b  = bc / CHUNKS;
    const int c  = bc - b * CHUNKS;
    const int m  = threadIdx.x;          // 0..511
    const float g = sigmoidf_dev(*ff);
    float* base = u + ((size_t)b * Ss + (size_t)c * CLEN) * Mm + m;
    float acc = 0.0f;
    for (int t = 0; t < CLEN; ++t) {
        acc = g * acc + base[(size_t)t * Mm];
        base[(size_t)t * Mm] = acc;
    }
    carry[((size_t)b * CHUNKS + c) * Mm + m] = acc;
}

// Pass B: exclusive scan of chunk carries with factor g^CLEN
__global__ __launch_bounds__(512)
void scan_carry_kernel(const float* __restrict__ carry, float* __restrict__ exc,
                       const float* __restrict__ ff) {
    const int b = blockIdx.x;
    const int m = threadIdx.x;
    const float g  = sigmoidf_dev(*ff);
    const float gL = powf(g, (float)CLEN);
    float p = 0.0f;
    for (int c = 0; c < CHUNKS; ++c) {
        const size_t o = ((size_t)b * CHUNKS + c) * Mm + m;
        exc[o] = p;
        p = gL * p + carry[o];
    }
}

// Pass C: mem[c*L + t] += g^{t+1} * exc[c]
__global__ __launch_bounds__(512)
void scan_fixup_kernel(float* __restrict__ u, const float* __restrict__ exc,
                       const float* __restrict__ ff) {
    const int bc = blockIdx.x;
    const int b  = bc / CHUNKS;
    const int c  = bc - b * CHUNKS;
    const int m  = threadIdx.x;
    const float g = sigmoidf_dev(*ff);
    const float e = exc[((size_t)b * CHUNKS + c) * Mm + m];
    float* base = u + ((size_t)b * Ss + (size_t)c * CLEN) * Mm + m;
    float f = g;
    for (int t = 0; t < CLEN; ++t) {
        base[(size_t)t * Mm] += f * e;
        f *= g;
    }
}

// ---------------------------------------------------------------------------
// retrieved = q * mem (elementwise)
// ---------------------------------------------------------------------------
__global__ __launch_bounds__(256)
void mul_kernel(const float* __restrict__ a, const float* __restrict__ b,
                float* __restrict__ o, size_t n) {
    size_t i = (size_t)blockIdx.x * blockDim.x + threadIdx.x;
    if (i < n) o[i] = a[i] * b[i];
}

// ---------------------------------------------------------------------------
// Launch
// ---------------------------------------------------------------------------
extern "C" void kernel_launch(void* const* d_in, const int* in_sizes, int n_in,
                              void* d_out, int out_size, void* d_ws, size_t ws_size,
                              hipStream_t stream) {
    const float* x   = (const float*)d_in[0];
    const float* Wd  = (const float*)d_in[1];
    const float* bd  = (const float*)d_in[2];
    const float* Wq  = (const float*)d_in[3];
    const float* bq  = (const float*)d_in[4];
    const float* Wk  = (const float*)d_in[5];
    const float* bk  = (const float*)d_in[6];
    const float* Wv  = (const float*)d_in[7];
    const float* bv  = (const float*)d_in[8];
    const float* gq  = (const float*)d_in[9];
    const float* bqln= (const float*)d_in[10];
    const float* gk  = (const float*)d_in[11];
    const float* bkln= (const float*)d_in[12];
    const float* W1  = (const float*)d_in[13];
    const float* W2  = (const float*)d_in[14];
    const float* Wu  = (const float*)d_in[15];
    const float* bu  = (const float*)d_in[16];
    const float* alr = (const float*)d_in[17];
    const float* ff  = (const float*)d_in[18];
    float* out = (float*)d_out;

    // ---- workspace carve-up ----------------------------------------------
    const size_t BSM = (size_t)ROWS * Mm;
    float* hbuf = (float*)d_ws;
    float* qbuf = hbuf + BSM;
    float* kbuf = qbuf + BSM;
    float* vbuf = kbuf + BSM;
    unsigned short* wt_d = (unsigned short*)(vbuf + BSM);          // [M][D] -> N=M,K=D
    unsigned short* wt_q = wt_d + (size_t)Dd * Mm;
    unsigned short* wt_k = wt_q + (size_t)Mm * Mm;
    unsigned short* wt_v = wt_k + (size_t)Mm * Mm;
    unsigned short* wt_1 = wt_v + (size_t)Mm * Mm;
    unsigned short* wt_2 = wt_1 + (size_t)Mm * Mm;
    unsigned short* wt_u = wt_2 + (size_t)Mm * Mm;                 // [D][M]
    float* carry = (float*)(wt_u + (size_t)Mm * Dd);
    float* exc   = carry + (size_t)Bb * CHUNKS * Mm;

    // ---- 0) convert+transpose weights to bf16 [N][K] ---------------------
    {
        int n1 = Dd * Mm;  // Wd, Wu
        int n2 = Mm * Mm;  // square ones
        convert_transpose_kernel<<<(n1 + 255) / 256, 256, 0, stream>>>(Wd, wt_d, Dd, Mm);
        convert_transpose_kernel<<<(n2 + 255) / 256, 256, 0, stream>>>(Wq, wt_q, Mm, Mm);
        convert_transpose_kernel<<<(n2 + 255) / 256, 256, 0, stream>>>(Wk, wt_k, Mm, Mm);
        convert_transpose_kernel<<<(n2 + 255) / 256, 256, 0, stream>>>(Wv, wt_v, Mm, Mm);
        convert_transpose_kernel<<<(n2 + 255) / 256, 256, 0, stream>>>(W1, wt_1, Mm, Mm);
        convert_transpose_kernel<<<(n2 + 255) / 256, 256, 0, stream>>>(W2, wt_2, Mm, Mm);
        convert_transpose_kernel<<<(n1 + 255) / 256, 256, 0, stream>>>(Wu, wt_u, Mm, Dd);
    }

    const dim3 blk(256);
    const dim3 grid_m(Mm / GBN, ROWS / GBM);   // N = 512
    const dim3 grid_d(Dd / GBN, ROWS / GBM);   // N = 1024

    // ---- 1) h = x @ Wd + bd ----------------------------------------------
    gemm_bf16_wmma<MODE_BIAS><<<grid_m, blk, 0, stream>>>(
        x, wt_d, bd, hbuf, Dd, Mm, nullptr, nullptr);

    // ---- 2) q = LN(h @ Wq + bq) ------------------------------------------
    gemm_bf16_wmma<MODE_BIAS><<<grid_m, blk, 0, stream>>>(
        hbuf, wt_q, bq, qbuf, Mm, Mm, nullptr, nullptr);
    layernorm_kernel<<<ROWS, 256, 0, stream>>>(qbuf, gq, bqln);

    // ---- 3) k = LN(h @ Wk + bk) ------------------------------------------
    gemm_bf16_wmma<MODE_BIAS><<<grid_m, blk, 0, stream>>>(
        hbuf, wt_k, bk, kbuf, Mm, Mm, nullptr, nullptr);
    layernorm_kernel<<<ROWS, 256, 0, stream>>>(kbuf, gk, bkln);

    // ---- 4) v = h @ Wv + bv ----------------------------------------------
    gemm_bf16_wmma<MODE_BIAS><<<grid_m, blk, 0, stream>>>(
        hbuf, wt_v, bv, vbuf, Mm, Mm, nullptr, nullptr);

    // ---- 5) t1 = gelu(k @ W1)   (h buffer reused) ------------------------
    gemm_bf16_wmma<MODE_GELU><<<grid_m, blk, 0, stream>>>(
        kbuf, wt_1, nullptr, hbuf, Mm, Mm, nullptr, nullptr);

    // ---- 6) u = lr * (v - t1 @ W2)  (in place in v buffer) ---------------
    gemm_bf16_wmma<MODE_U><<<grid_m, blk, 0, stream>>>(
        hbuf, wt_2, nullptr, vbuf, Mm, Mm, vbuf, alr);

    // ---- 7) gated linear recurrence (blocked scan, in place on v) --------
    scan_local_kernel<<<Bb * CHUNKS, 512, 0, stream>>>(vbuf, carry, ff);
    scan_carry_kernel<<<Bb, 512, 0, stream>>>(carry, exc, ff);
    scan_fixup_kernel<<<Bb * CHUNKS, 512, 0, stream>>>(vbuf, exc, ff);

    // ---- 8) retrieved = q * mem  (k buffer reused) -----------------------
    {
        size_t n = BSM;
        mul_kernel<<<(unsigned)((n + 255) / 256), 256, 0, stream>>>(qbuf, vbuf, kbuf, n);
    }

    // ---- 9) out = x + retrieved @ Wu + bu --------------------------------
    gemm_bf16_wmma<MODE_RESID><<<grid_d, blk, 0, stream>>>(
        kbuf, wt_u, bu, out, Mm, Dd, x, nullptr);

    (void)in_sizes; (void)n_in; (void)out_size; (void)ws_size;
}
